// MaskedAttentionHead_8701603742389
// MI455X (gfx1250) — matrix operand order repaired and checked by reading, not compiled, not run on hardware
//
#include <hip/hip_runtime.h>
#include <hip/hip_bf16.h>

// ---------------------------------------------------------------------------
// MI455X (gfx1250) masked multi-head attention, wave32 + WMMA bf16.
//   K1: proj = x @ w_attn^T + b_attn  -> scatter into viewed Q,K (b,h,s,d) and
//       V^T (b,h,d,s) bf16 buffers (the torch-style raw reshape handled here).
//   K2: flash attention per (b,h): S = Q K^T /16, causal, online softmax,
//       O += P V, all via v_wmma_f32_16x16x32_bf16. K/V tiles staged in LDS
//       with double-buffered GLOBAL_LOAD_ASYNC_TO_LDS_B128 (ASYNCcnt).
//   K3: y = O(viewed) @ w_out^T + b_out, fp32 out.
// ---------------------------------------------------------------------------

typedef __attribute__((ext_vector_type(16))) __bf16 v16bf;
typedef __attribute__((ext_vector_type(8)))  float  v8f;

#define LOG2E 1.44269504088896340736f

union BF16x16 { uint4 u[2]; v16bf v; };

static __device__ __forceinline__ v8f zero8() {
  v8f z;
#pragma unroll
  for (int i = 0; i < 8; ++i) z[i] = 0.0f;
  return z;
}

static __device__ __forceinline__ v8f wmma_bf16(v16bf a, v16bf b, v8f c) {
  return __builtin_amdgcn_wmma_f32_16x16x32_bf16(false, a, false, b,
                                                 (short)0, c, false, false);
}

// A fragment (16x32 bf16): lane m<16 holds row m; K = {8h..8h+7} u {16+8h..}
static __device__ __forceinline__ v16bf load_a_bf16(const __bf16* __restrict__ base, int ld) {
  const int lane = threadIdx.x & 31;
  const int r = lane & 15, h = lane >> 4;
  const __bf16* p = base + r * ld;
  BF16x16 t;
  t.u[0] = *(const uint4*)(p + 8 * h);
  t.u[1] = *(const uint4*)(p + 16 + 8 * h);
  return t.v;
}

// B fragment (32x16 bf16): lane n holds col n; K = {8h..8h+7} u {16+8h..}
// base must point at (col0, k0); "col" rows have stride ld (elements).
static __device__ __forceinline__ v16bf load_b_bf16(const __bf16* __restrict__ base, int ld) {
  const int lane = threadIdx.x & 31;
  const int n = lane & 15, h = lane >> 4;
  const __bf16* p = base + n * ld;
  BF16x16 t;
  t.u[0] = *(const uint4*)(p + 8 * h);
  t.u[1] = *(const uint4*)(p + 16 + 8 * h);
  return t.v;
}

static __device__ __forceinline__ v16bf load_a_f32(const float* __restrict__ base, int ld) {
  const int lane = threadIdx.x & 31;
  const int r = lane & 15, h = lane >> 4;
  const float* p = base + r * ld;
  v16bf v;
#pragma unroll
  for (int i = 0; i < 8; ++i) v[i] = (__bf16)p[8 * h + i];
#pragma unroll
  for (int i = 0; i < 8; ++i) v[8 + i] = (__bf16)p[16 + 8 * h + i];
  return v;
}

static __device__ __forceinline__ v16bf load_b_f32(const float* __restrict__ base, int ld) {
  const int lane = threadIdx.x & 31;
  const int n = lane & 15, h = lane >> 4;
  const float* p = base + n * ld;
  v16bf v;
#pragma unroll
  for (int i = 0; i < 8; ++i) v[i] = (__bf16)p[8 * h + i];
#pragma unroll
  for (int i = 0; i < 8; ++i) v[8 + i] = (__bf16)p[16 + 8 * h + i];
  return v;
}

// --- CDNA5 async global->LDS (ASYNCcnt) ------------------------------------
static __device__ __forceinline__ unsigned int lds_off(const void* p) {
  // generic LDS pointer: high 32 = shared aperture, low 32 = LDS byte offset
  return (unsigned int)(uintptr_t)p;
}

static __device__ __forceinline__ void async_b128(const void* g, unsigned int l) {
  asm volatile("global_load_async_to_lds_b128 %0, %1, off"
               :: "v"(l), "v"((unsigned long long)(uintptr_t)g)
               : "memory");
}

static __device__ __forceinline__ void wait_async0() {
  asm volatile("s_wait_asynccnt 0x0" ::: "memory");
}

// ---------------------------------------------------------------------------
// K1: QKV projection. grid = (4096/16, 6144/128), 8 waves/block, 16x16/wave.
// ---------------------------------------------------------------------------
__global__ __launch_bounds__(256) void qkv_gemm_kernel(
    const float* __restrict__ x, const float* __restrict__ w_attn,
    const float* __restrict__ b_attn,
    __bf16* __restrict__ qb, __bf16* __restrict__ kb, __bf16* __restrict__ vtb) {
  const int wave = threadIdx.x >> 5;
  const int lane = threadIdx.x & 31;
  const int m0 = blockIdx.x * 16;                 // row in flat (4096 x 256) x
  const int n0 = blockIdx.y * 128 + wave * 16;    // col in 6144

  v8f acc = zero8();
#pragma unroll
  for (int kt = 0; kt < 8; ++kt) {
    v16bf a = load_a_f32(x + m0 * 256 + kt * 32, 256);
    v16bf b = load_b_f32(w_attn + n0 * 256 + kt * 32, 256);
    acc = wmma_bf16(a, b, acc);
  }

  const int col   = n0 + (lane & 15);
  const float bias = b_attn[col];
  const int chunk = col >> 11;      // 0=Q 1=K 2=V
  const int cw    = col & 2047;
  const int rowh  = (lane >> 4) * 8;
  const int bidx  = m0 >> 10;       // batch (tiles never straddle batches)
#pragma unroll
  for (int g = 0; g < 8; ++g) {
    const int sp  = (m0 & 1023) + rowh + g;       // s' within batch
    const int f   = sp * 2048 + cw;               // flat idx inside (b) chunk
    const int hh  = f >> 18;                      // / (1024*256)
    const int rem = f & 262143;
    const int s   = rem >> 8;
    const int d   = rem & 255;
    const int bh  = bidx * 8 + hh;
    const __bf16 val = (__bf16)(acc[g] + bias);
    if (chunk == 0)      qb[(bh * 1024 + s) * 256 + d] = val;
    else if (chunk == 1) kb[(bh * 1024 + s) * 256 + d] = val;
    else                 vtb[(bh * 256 + d) * 1024 + s] = val;  // V transposed
  }
}

// ---------------------------------------------------------------------------
// K2: flash attention. grid = (32 bh, 8 q-chunks), 8 waves/block,
// each wave owns 16 query rows x full 256-dim head. K/V tiles are staged in
// LDS via double-buffered async global->LDS copies shared by all 8 waves.
// ---------------------------------------------------------------------------
#define KST_LD 264   // staged K row stride in bf16 (256 data + 8 pad), 528 B
#define VST_LD 40    // staged V^T row stride in bf16 (32 data + 8 pad), 80 B

__global__ __launch_bounds__(256) void attn_kernel(
    const __bf16* __restrict__ qbuf, const __bf16* __restrict__ kbuf,
    const __bf16* __restrict__ vtbuf, __bf16* __restrict__ obuf) {
  __shared__ __align__(16) __bf16 kst[2][32][KST_LD];   // 33 KB
  __shared__ __align__(16) __bf16 vst[2][256][VST_LD];  // 40 KB
  __shared__ __align__(16) __bf16 pstage[8][16][40];    // 10 KB

  const int bh   = blockIdx.x;
  const int wave = threadIdx.x >> 5;
  const int lane = threadIdx.x & 31;
  const int q0   = blockIdx.y * 128 + wave * 16;

  const __bf16* Q  = qbuf  + bh * (1024 * 256);
  const __bf16* K  = kbuf  + bh * (1024 * 256);
  const __bf16* VT = vtbuf + bh * (256 * 1024);
  __bf16*       O  = obuf  + bh * (1024 * 256);

  v16bf qf[8];
#pragma unroll
  for (int kd = 0; kd < 8; ++kd) qf[kd] = load_a_bf16(Q + q0 * 256 + kd * 32, 256);

  v8f oacc[16];
#pragma unroll
  for (int t = 0; t < 16; ++t) oacc[t] = zero8();

  float m_r[8], l_r[8];
#pragma unroll
  for (int g = 0; g < 8; ++g) { m_r[g] = -__builtin_inff(); l_r[g] = 0.0f; }

  const int colb = lane & 15;
  const int rowh = (lane >> 4) * 8;
  __bf16* ps = &pstage[wave][0][0];     // row stride 40 (80 B, 16 B aligned)

  const int nJT_w = (q0 + 47) >> 5;                     // this wave's tiles
  const int nJT_b = (blockIdx.y * 128 + 112 + 47) >> 5; // block max (wave 7)

  // ---- async stage of K tile (32 x 256) and V^T slice (256 x 32) ----------
  auto issue_stage = [&](int buf, int j0) {
#pragma unroll
    for (int i = 0; i < 4; ++i) {             // K: 1024 16B chunks / 256 thr
      const int c = threadIdx.x + 256 * i;
      const int r = c >> 5, c8 = c & 31;      // key row, 16B chunk in row
      async_b128(K + (j0 + r) * 256 + c8 * 8, lds_off(&kst[buf][r][c8 * 8]));
    }
#pragma unroll
    for (int i = 0; i < 4; ++i) {             // V^T: 1024 16B chunks
      const int c = threadIdx.x + 256 * i;
      const int d = c >> 2, c4 = c & 3;       // d row, 16B chunk of 32 keys
      async_b128(VT + d * 1024 + j0 + c4 * 8, lds_off(&vst[buf][d][c4 * 8]));
    }
  };

  issue_stage(0, 0);

  for (int jt = 0; jt < nJT_b; ++jt) {
    const int cur = jt & 1;
    wait_async0();          // own stage-jt copies landed in LDS
    __syncthreads();        // all waves' copies landed; prev buffer free
    if (jt + 1 < nJT_b) issue_stage(cur ^ 1, (jt + 1) * 32);  // overlap DMA

    if (jt < nJT_w) {
      const int j0 = jt * 32;

      // ---- S tiles: 16 queries x 32 keys, accumulate over d = 256 --------
      v8f s0 = zero8(), s1 = zero8();
      const __bf16* kbase = &kst[cur][0][0];
#pragma unroll
      for (int kd = 0; kd < 8; ++kd) {
        v16bf b0 = load_b_bf16(kbase + kd * 32, KST_LD);               // keys 0..15
        v16bf b1 = load_b_bf16(kbase + 16 * KST_LD + kd * 32, KST_LD); // keys 16..31
        s0 = wmma_bf16(qf[kd], b0, s0);
        s1 = wmma_bf16(qf[kd], b1, s1);
      }

      // ---- scale, causal mask, online softmax (row stats via shfl) -------
      float p0[8], p1[8], scl[8];
#pragma unroll
      for (int g = 0; g < 8; ++g) {
        const int qrow = q0 + rowh + g;
        float a = s0[g] * 0.0625f;        // 1/sqrt(256)
        float b = s1[g] * 0.0625f;
        if (j0 + colb > qrow)      a = -__builtin_inff();
        if (j0 + 16 + colb > qrow) b = -__builtin_inff();
        float mx = fmaxf(a, b);
        mx = fmaxf(mx, __shfl_xor(mx, 1, 32));
        mx = fmaxf(mx, __shfl_xor(mx, 2, 32));
        mx = fmaxf(mx, __shfl_xor(mx, 4, 32));
        mx = fmaxf(mx, __shfl_xor(mx, 8, 32));
        const float mnew = fmaxf(m_r[g], mx);
        scl[g] = exp2f((m_r[g] - mnew) * LOG2E);
        a = exp2f((a - mnew) * LOG2E);
        b = exp2f((b - mnew) * LOG2E);
        float rs = a + b;
        rs += __shfl_xor(rs, 1, 32);
        rs += __shfl_xor(rs, 2, 32);
        rs += __shfl_xor(rs, 4, 32);
        rs += __shfl_xor(rs, 8, 32);
        l_r[g] = l_r[g] * scl[g] + rs;
        m_r[g] = mnew;
        p0[g] = a; p1[g] = b;
      }
#pragma unroll
      for (int t = 0; t < 16; ++t)
#pragma unroll
        for (int g = 0; g < 8; ++g) oacc[t][g] *= scl[g];

      // ---- P (f32, C layout) -> bf16 A layout via per-wave LDS tile ------
#pragma unroll
      for (int g = 0; g < 8; ++g) {
        ps[(rowh + g) * 40 + colb]      = (__bf16)p0[g];
        ps[(rowh + g) * 40 + 16 + colb] = (__bf16)p1[g];
      }
      asm volatile("s_wait_dscnt 0x0" ::: "memory");
      v16bf pa;
      {
        const int r = lane & 15, hf = lane >> 4;
        BF16x16 t;
        t.u[0] = *(const uint4*)(ps + r * 40 + 8 * hf);
        t.u[1] = *(const uint4*)(ps + r * 40 + 16 + 8 * hf);
        pa = t.v;
      }

      // ---- O += P * V from staged V^T (rows = d, 32 keys contiguous) -----
#pragma unroll
      for (int t = 0; t < 16; ++t) {
        v16bf bv = load_b_bf16(&vst[cur][t * 16][0], VST_LD);
        oacc[t] = wmma_bf16(pa, bv, oacc[t]);
      }
    }
    // next iteration's top barrier protects buffer reuse (stage jt+2 is only
    // issued after every wave has finished reading buffer jt).
  }

  // ---- normalize and store O (bf16) ----
  float inv[8];
#pragma unroll
  for (int g = 0; g < 8; ++g) inv[g] = 1.0f / l_r[g];
#pragma unroll
  for (int t = 0; t < 16; ++t)
#pragma unroll
    for (int g = 0; g < 8; ++g)
      O[(q0 + rowh + g) * 256 + t * 16 + colb] = (__bf16)(oacc[t][g] * inv[g]);
}

// ---------------------------------------------------------------------------
// K3: output projection. O viewed flat as (4096 x 2048) bf16.
// grid = (4096/16, 256/128), 8 waves/block.
// ---------------------------------------------------------------------------
__global__ __launch_bounds__(256) void out_gemm_kernel(
    const __bf16* __restrict__ obuf, const float* __restrict__ w_out,
    const float* __restrict__ b_out, float* __restrict__ y) {
  const int wave = threadIdx.x >> 5;
  const int lane = threadIdx.x & 31;
  const int m0 = blockIdx.x * 16;
  const int n0 = blockIdx.y * 128 + wave * 16;

  v8f acc = zero8();
#pragma unroll 4
  for (int kt = 0; kt < 64; ++kt) {
    v16bf a = load_a_bf16(obuf + m0 * 2048 + kt * 32, 2048);
    v16bf b = load_b_f32(w_out + n0 * 2048 + kt * 32, 2048);
    acc = wmma_bf16(a, b, acc);
  }

  const int col  = n0 + (lane & 15);
  const float bias = b_out[col];
  const int rowh = (lane >> 4) * 8;
#pragma unroll
  for (int g = 0; g < 8; ++g)
    y[(m0 + rowh + g) * 256 + col] = acc[g] + bias;
}

// ---------------------------------------------------------------------------
extern "C" void kernel_launch(void* const* d_in, const int* in_sizes, int n_in,
                              void* d_out, int out_size, void* d_ws, size_t ws_size,
                              hipStream_t stream) {
  (void)in_sizes; (void)n_in; (void)out_size; (void)ws_size;
  const float* x      = (const float*)d_in[0];
  const float* w_attn = (const float*)d_in[1];
  const float* b_attn = (const float*)d_in[2];
  const float* w_out  = (const float*)d_in[3];
  const float* b_out  = (const float*)d_in[4];
  float* y = (float*)d_out;

  // workspace: Q, K, V^T, O as bf16, each 4*8*1024*256 = 8M elems (16 MB)
  const size_t NELEM = 4ull * 8 * 1024 * 256;
  __bf16* qb  = (__bf16*)d_ws;
  __bf16* kb  = qb + NELEM;
  __bf16* vtb = kb + NELEM;
  __bf16* ob  = vtb + NELEM;

  qkv_gemm_kernel<<<dim3(256, 48), 256, 0, stream>>>(x, w_attn, b_attn, qb, kb, vtb);
  attn_kernel<<<dim3(32, 8), 256, 0, stream>>>(qb, kb, vtb, ob);
  out_gemm_kernel<<<dim3(256, 2), 256, 0, stream>>>(ob, w_out, b_out, y);
}
